// MultiHeadAttention_59141699666069
// MI455X (gfx1250) — matrix operand order repaired and checked by reading, not compiled, not run on hardware
//
#include <hip/hip_runtime.h>
#include <hip/hip_bf16.h>

// ---------------------------------------------------------------------------
// MultiHeadAttention for MI455X (gfx1250): f16 WMMA everywhere, flash-style
// attention (no [B,H,S,S] materialization), fp32 accumulation + softmax.
// ---------------------------------------------------------------------------

typedef __attribute__((ext_vector_type(16))) _Float16 v16h;
typedef __attribute__((ext_vector_type(8)))  _Float16 v8h;
typedef __attribute__((ext_vector_type(8)))  float    v8f;

#define NB  2
#define NS  2048
#define ND  1024
#define NH  16
#define NDH 64
#define NM  (NB * NS)   // 4096 rows

// ---- WMMA wrapper ----------------------------------------------------------
__device__ __forceinline__ v8f wmma16(v16h a, v16h b, v8f c) {
  return __builtin_amdgcn_wmma_f32_16x16x32_f16(false, a, false, b,
                                                (short)0, c, false, false);
}

// ---- fragment loaders ------------------------------------------------------
// A operand (16x32 f16, MxK): lane holds row M=lane&15; K chunks
// [hh, hh+8) and [16+hh, 16+hh+8) with hh = (lane>>4)*8.
// Caller passes p = row_base + hh; we read p[0..7] and p[16..23].
__device__ __forceinline__ v16h load_a_h(const _Float16* p) {
  v8h lo = *(const v8h*)(p);
  v8h hi = *(const v8h*)(p + 16);
  v16h r;
#pragma unroll
  for (int i = 0; i < 8; ++i) { r[i] = lo[i]; r[i + 8] = hi[i]; }
  return r;
}

// Same A layout but source is fp32 (converted to f16 in-register).
__device__ __forceinline__ v16h load_a_f32(const float* p) {
  v16h r;
#pragma unroll
  for (int i = 0; i < 8; ++i) {
    r[i]     = (_Float16)p[i];
    r[i + 8] = (_Float16)p[16 + i];
  }
  return r;
}

// B operand (32x16 f16, KxN): lane holds column N=lane&15; K chunks
// [hh, hh+8) and [16+hh, 16+hh+8) with hh = (lane>>4)*8. Caller passes
// p = col_base + hh; we read p[0..7] and p[16..23].
__device__ __forceinline__ v16h load_b_h(const _Float16* p) {
  v8h lo = *(const v8h*)(p);
  v8h hi = *(const v8h*)(p + 16);
  v16h r;
#pragma unroll
  for (int i = 0; i < 8; ++i) { r[i] = lo[i]; r[i + 8] = hi[i]; }
  return r;
}

// ---------------------------------------------------------------------------
// Kernel 1: weight transpose + fp32->fp16.
// mode 0: W is [H, D, DH] stacked per-head dense  -> WT[n*K + d], n = h*64+e
// mode 1: W is [D, D] row-major (Wo)              -> WT[n*K + d] = W[d*D + n]
// ---------------------------------------------------------------------------
__global__ __launch_bounds__(256) void xpose_w(const float* __restrict__ W,
                                               _Float16* __restrict__ WT,
                                               int mode) {
  int idx = blockIdx.x * 256 + threadIdx.x;     // [0, 1024*1024)
  int n = idx >> 10, d = idx & 1023;
  float v = mode ? W[d * ND + n]
                 : W[(((n >> 6) * ND) + d) * NDH + (n & 63)];
  WT[idx] = (_Float16)v;
}

// ---------------------------------------------------------------------------
// Kernel 2: projection GEMM  C[M,N] = f16(X[M,K]) * WT^T + bias
// X fp32 row-major [M,1024]; WT fp16 [N][K] (K contiguous); bias fp32 [N].
// vmode 0: out fp16 row-major [M, 1024]   (used for q, k)
// vmode 1: out fp16 transposed per head [B, H, DH, S]  (used for v)
// Each wave computes a 32x32 tile (2x2 WMMA tiles). 8 waves/block.
// ---------------------------------------------------------------------------
__global__ __launch_bounds__(256) void proj_gemm(const float* __restrict__ X,
                                                 const _Float16* __restrict__ WT,
                                                 const float* __restrict__ bias,
                                                 _Float16* __restrict__ out,
                                                 int vmode) {
  constexpr int K = ND, N = ND;
  const int lane = threadIdx.x & 31;
  const int w    = blockIdx.x * 8 + (threadIdx.x >> 5);
  const int tm = w >> 5;          // M/32 = 128 tiles
  const int tn = w & 31;          // N/32 = 32 tiles
  const int ci  = lane & 15;
  const int hh  = (lane >> 4) << 3;   // 0 or 8  (A chunk / C row offset)
  const int kh  = (lane >> 4) << 3;   // 0 or 8 (B chunk K offset)

  const float*    xa0 = X  + (size_t)(tm * 32 + ci)      * K + hh;
  const float*    xa1 = X  + (size_t)(tm * 32 + 16 + ci) * K + hh;
  const _Float16* wb0 = WT + (size_t)(tn * 32 + ci)      * K + kh;
  const _Float16* wb1 = WT + (size_t)(tn * 32 + 16 + ci) * K + kh;

  v8f acc[2][2] = {};
  for (int kk = 0; kk < K; kk += 32) {
    v16h a0 = load_a_f32(xa0 + kk);
    v16h a1 = load_a_f32(xa1 + kk);
    v16h b0 = load_b_h(wb0 + kk);
    v16h b1 = load_b_h(wb1 + kk);
    acc[0][0] = wmma16(a0, b0, acc[0][0]);
    acc[0][1] = wmma16(a0, b1, acc[0][1]);
    acc[1][0] = wmma16(a1, b0, acc[1][0]);
    acc[1][1] = wmma16(a1, b1, acc[1][1]);
  }

#pragma unroll
  for (int j = 0; j < 2; ++j) {
    const int n  = tn * 32 + j * 16 + ci;   // lane's C column
    const float bv = bias[n];
#pragma unroll
    for (int i = 0; i < 2; ++i) {
      const int m0 = tm * 32 + i * 16 + hh; // lane's first C row (8 rows total)
      if (!vmode) {
#pragma unroll
        for (int r = 0; r < 8; ++r)
          out[(size_t)(m0 + r) * N + n] = (_Float16)(acc[i][j][r] + bv);
      } else {
        // v transposed: vt[((b*H + h)*DH + dh) * S + s], rows r are
        // consecutive s -> one contiguous 16B store per lane.
        const int bb = m0 >> 11, s0 = m0 & (NS - 1);
        const int hd = n >> 6, dh = n & 63;
        v8h pack;
#pragma unroll
        for (int r = 0; r < 8; ++r) pack[r] = (_Float16)(acc[i][j][r] + bv);
        *(v8h*)(out + ((size_t)((bb * NH + hd) * NDH + dh)) * NS + s0) = pack;
      }
    }
  }
}

// ---------------------------------------------------------------------------
// Kernel 3: flash attention.
// q,k fp16 [B,S,D] (head slice h*64..), vT fp16 [B,H,DH,S], mask int32 [B,S,S].
// Each wave owns 16 query rows; streams keys 32 at a time with online softmax.
// Output: fp16 [B,S,D].
// ---------------------------------------------------------------------------
__global__ __launch_bounds__(256) void flash_attn(const _Float16* __restrict__ qb,
                                                  const _Float16* __restrict__ kb,
                                                  const _Float16* __restrict__ vt,
                                                  const int* __restrict__ mask,
                                                  _Float16* __restrict__ attn) {
  __shared__ _Float16 plds[8 * 16 * 32];        // per-wave 16x32 P staging

  const int lane = threadIdx.x & 31;
  const int widx = threadIdx.x >> 5;
  const int qt = blockIdx.x & 15;               // 16 q-tiles of 128 rows
  const int h  = (blockIdx.x >> 4) & 15;
  const int b  = blockIdx.x >> 8;
  const int q0 = qt * 128 + widx * 16;

  const int ci = lane & 15;
  const int hh = (lane >> 4) << 3;              // A chunk / C row offset
  const int kh = (lane >> 4) << 3;              // B chunk K offset

  // Q fragments (A operand), resident for the whole loop. K dim = dh (64).
  const _Float16* qrow = qb + ((size_t)(b * NS + q0 + ci)) * ND + h * NDH;
  const v16h aq0 = load_a_h(qrow + hh);         // dh 0..31
  const v16h aq1 = load_a_h(qrow + 32 + hh);    // dh 32..63

  float mrow[8], lrow[8];
  v8f o[4] = {};
#pragma unroll
  for (int r = 0; r < 8; ++r) { mrow[r] = -3.0e38f; lrow[r] = 0.f; }

  _Float16* wp = plds + widx * 512;

  for (int kt = 0; kt < NS; kt += 32) {
    // ---- logits: two 16x16 tiles over 32 keys, K-dim = dh = 64 (2 WMMA ea)
    v8f sl[2];
#pragma unroll
    for (int jt = 0; jt < 2; ++jt) {
      const _Float16* krow =
          kb + ((size_t)(b * NS + kt + jt * 16 + ci)) * ND + h * NDH;
      v16h bk0 = load_b_h(krow + kh);           // dh 0..31 for this lane half
      v16h bk1 = load_b_h(krow + 32 + kh);      // dh 32..63
      v8f s = {};
      s = wmma16(aq0, bk0, s);
      s = wmma16(aq1, bk1, s);
      sl[jt] = s;
    }
    if (kt + 32 < NS)
      __builtin_prefetch(kb + ((size_t)(b * NS + kt + 32 + ci)) * ND + h * NDH, 0, 0);

    // ---- scale + additive mask (mask==1 -> -1e9)
    const int* mp = mask + ((size_t)(b * NS) + q0 + hh) * NS + kt + ci;
#pragma unroll
    for (int r = 0; r < 8; ++r) {
      int mv0 = mp[r * NS];
      int mv1 = mp[r * NS + 16];
      sl[0][r] = sl[0][r] * 0.125f + (mv0 ? -1.0e9f : 0.f);
      sl[1][r] = sl[1][r] * 0.125f + (mv1 ? -1.0e9f : 0.f);
    }

    // ---- online softmax statistics (reduce across 16 lanes of each half)
    float corr[8], mnew[8];
#pragma unroll
    for (int r = 0; r < 8; ++r) {
      float x = fmaxf(sl[0][r], sl[1][r]);
      x = fmaxf(x, __shfl_xor(x, 1, 32));
      x = fmaxf(x, __shfl_xor(x, 2, 32));
      x = fmaxf(x, __shfl_xor(x, 4, 32));
      x = fmaxf(x, __shfl_xor(x, 8, 32));
      mnew[r] = fmaxf(mrow[r], x);
      corr[r] = __expf(mrow[r] - mnew[r]);
      mrow[r] = mnew[r];
    }
#pragma unroll
    for (int r = 0; r < 8; ++r) {
      sl[0][r] = __expf(sl[0][r] - mnew[r]);
      sl[1][r] = __expf(sl[1][r] - mnew[r]);
      float s = sl[0][r] + sl[1][r];
      s += __shfl_xor(s, 1, 32);
      s += __shfl_xor(s, 2, 32);
      s += __shfl_xor(s, 4, 32);
      s += __shfl_xor(s, 8, 32);
      lrow[r] = lrow[r] * corr[r] + s;
    }

    // ---- stage P (C layout) -> LDS -> reload as A fragment (layout xform)
#pragma unroll
    for (int r = 0; r < 8; ++r) {
      wp[(hh + r) * 32 + ci]      = (_Float16)sl[0][r];
      wp[(hh + r) * 32 + 16 + ci] = (_Float16)sl[1][r];
    }
    asm volatile("s_wait_dscnt 0x0" ::: "memory");
    v16h pa = load_a_h(wp + ci * 32 + hh);

    // ---- rescale O, then O += P(16x32) x V(32x64): 4 WMMA over dh tiles
#pragma unroll
    for (int t = 0; t < 4; ++t) {
#pragma unroll
      for (int r = 0; r < 8; ++r) o[t][r] *= corr[r];
      const _Float16* vrow =
          vt + ((size_t)((b * NH + h) * NDH + t * 16 + ci)) * NS + kt + kh;
      v16h vb = load_b_h(vrow);
      o[t] = wmma16(pa, vb, o[t]);
    }
  }

  // ---- normalize and write attn output (fp16, row-major [B,S,D])
#pragma unroll
  for (int r = 0; r < 8; ++r) {
    const float inv = 1.0f / lrow[r];
    const size_t row = (size_t)(b * NS + q0 + hh + r) * ND + h * NDH;
#pragma unroll
    for (int t = 0; t < 4; ++t)
      attn[row + t * 16 + ci] = (_Float16)(o[t][r] * inv);
  }
}

// ---------------------------------------------------------------------------
// Kernel 4: output projection  out[M,N] = attn(f16) * WoT^T + bo, fp32 out.
// ---------------------------------------------------------------------------
__global__ __launch_bounds__(256) void out_gemm(const _Float16* __restrict__ A,
                                                const _Float16* __restrict__ WT,
                                                const float* __restrict__ bias,
                                                float* __restrict__ out) {
  constexpr int K = ND, N = ND;
  const int lane = threadIdx.x & 31;
  const int w    = blockIdx.x * 8 + (threadIdx.x >> 5);
  const int tm = w >> 5, tn = w & 31;
  const int ci = lane & 15;
  const int hh = (lane >> 4) << 3;
  const int kh = (lane >> 4) << 3;

  const _Float16* xa0 = A  + (size_t)(tm * 32 + ci)      * K + hh;
  const _Float16* xa1 = A  + (size_t)(tm * 32 + 16 + ci) * K + hh;
  const _Float16* wb0 = WT + (size_t)(tn * 32 + ci)      * K + kh;
  const _Float16* wb1 = WT + (size_t)(tn * 32 + 16 + ci) * K + kh;

  v8f acc[2][2] = {};
  for (int kk = 0; kk < K; kk += 32) {
    v16h a0 = load_a_h(xa0 + kk);
    v16h a1 = load_a_h(xa1 + kk);
    v16h b0 = load_b_h(wb0 + kk);
    v16h b1 = load_b_h(wb1 + kk);
    acc[0][0] = wmma16(a0, b0, acc[0][0]);
    acc[0][1] = wmma16(a0, b1, acc[0][1]);
    acc[1][0] = wmma16(a1, b0, acc[1][0]);
    acc[1][1] = wmma16(a1, b1, acc[1][1]);
  }

#pragma unroll
  for (int j = 0; j < 2; ++j) {
    const int n  = tn * 32 + j * 16 + ci;
    const float bv = bias[n];
#pragma unroll
    for (int i = 0; i < 2; ++i) {
      const int m0 = tm * 32 + i * 16 + hh;
#pragma unroll
      for (int r = 0; r < 8; ++r)
        out[(size_t)(m0 + r) * N + n] = acc[i][j][r] + bv;
    }
  }
}

// ---------------------------------------------------------------------------
extern "C" void kernel_launch(void* const* d_in, const int* in_sizes, int n_in,
                              void* d_out, int out_size, void* d_ws, size_t ws_size,
                              hipStream_t stream) {
  const float* query = (const float*)d_in[0];
  const float* key   = (const float*)d_in[1];
  const float* value = (const float*)d_in[2];
  const int*   mask  = (const int*)  d_in[3];
  const float* Wq    = (const float*)d_in[4];
  const float* bq    = (const float*)d_in[5];
  const float* Wk    = (const float*)d_in[6];
  const float* bk    = (const float*)d_in[7];
  const float* Wv    = (const float*)d_in[8];
  const float* bv    = (const float*)d_in[9];
  const float* Wo    = (const float*)d_in[10];
  const float* bo    = (const float*)d_in[11];
  float* out = (float*)d_out;

  // workspace layout (halves): 4x 1M weight^T + 4x 4M activations = 20M halves
  _Float16* ws  = (_Float16*)d_ws;
  _Float16* wqT = ws;
  _Float16* wkT = wqT + (size_t)ND * ND;
  _Float16* wvT = wkT + (size_t)ND * ND;
  _Float16* woT = wvT + (size_t)ND * ND;
  _Float16* qh  = woT + (size_t)ND * ND;
  _Float16* kh  = qh  + (size_t)NM * ND;
  _Float16* vth = kh  + (size_t)NM * ND;
  _Float16* ah  = vth + (size_t)NM * ND;

  const dim3 blk(256);
  const int xg = (ND * ND) / 256;               // 4096
  hipLaunchKernelGGL(xpose_w, dim3(xg), blk, 0, stream, Wq, wqT, 0);
  hipLaunchKernelGGL(xpose_w, dim3(xg), blk, 0, stream, Wk, wkT, 0);
  hipLaunchKernelGGL(xpose_w, dim3(xg), blk, 0, stream, Wv, wvT, 0);
  hipLaunchKernelGGL(xpose_w, dim3(xg), blk, 0, stream, Wo, woT, 1);

  const int gg = (NM / 32) * (ND / 32) / 8;     // 512 blocks
  hipLaunchKernelGGL(proj_gemm, dim3(gg), blk, 0, stream, query, wqT, bq, qh, 0);
  hipLaunchKernelGGL(proj_gemm, dim3(gg), blk, 0, stream, key,   wkT, bk, kh, 0);
  hipLaunchKernelGGL(proj_gemm, dim3(gg), blk, 0, stream, value, wvT, bv, vth, 1);

  hipLaunchKernelGGL(flash_attn, dim3(NB * NH * (NS / 128)), blk, 0, stream,
                     qh, kh, vth, mask, ah);

  hipLaunchKernelGGL(out_gemm, dim3(gg), blk, 0, stream, ah, woT, bo, out);
}
